// GCNNet_9320079033239
// MI455X (gfx1250) — matrix operand order repaired and checked by reading, not compiled, not run on hardware
//
#include <hip/hip_runtime.h>
#include <hip/hip_bf16.h>
#include <stdint.h>

// ---------------------------------------------------------------------------
// GCNNet forward for MI455X (gfx1250, wave32, WMMA)
//   h  = emb[x].reshape(8000, 4000)
//   h1 = leaky(agg(h  @ W1) + b1)        agg = D^-1/2 (A+I) D^-1/2
//   h2 = leaky(agg(h1 @ W2) + b2)
//   g  = segment_max(h2, batch)          [64, 6000]
//   g1 = leaky(g @ Wl1 + bl1)            [64, 3000]
//   out= leaky(g1 @ Wl2 + bl2)           [64, 4]
// GEMMs run in bf16 via v_wmma_f32_16x16x32_bf16 (f32 accumulate).
// ---------------------------------------------------------------------------

typedef __attribute__((ext_vector_type(16))) __bf16 v16bf;
typedef __attribute__((ext_vector_type(8)))  float  v8f;

#define LRELU 0.01f

// Problem constants (from reference setup_inputs)
#define NN   8000
#define EE   32000
#define BB   64
#define DD   200
#define TT   20
#define K1   4000          // conv1 in-dim
#define H1   6000          // hidden dim
#define H1P  6016          // padded to multiple of 64 (N-tiles) and 32 (K-steps)
#define L1N  3000          // lin1 out-dim
#define L1NP 3008          // padded

static __device__ __forceinline__ __bf16 f2bf(float f) {
  union { float f; uint32_t u; } in; in.f = f;
  uint32_t u = in.u;
  uint32_t r = (u + 0x7FFFu + ((u >> 16) & 1u)) >> 16;   // round-to-nearest-even
  uint16_t s = (uint16_t)r;
  __bf16 b; __builtin_memcpy(&b, &s, 2);
  return b;
}

static __device__ __forceinline__ float leaky(float v) {
  return v > 0.0f ? v : LRELU * v;
}

// ------------------------------- utility -----------------------------------

__global__ void fill_f32(float* __restrict__ p, long long n, float v) {
  long long i = (long long)blockIdx.x * blockDim.x + threadIdx.x;
  if (i < n) p[i] = v;
}

// degree accumulation over edge destinations (self-loops pre-seeded as 1.0)
__global__ void deg_accum(const int* __restrict__ dst, int E, float* __restrict__ deg) {
  int e = blockIdx.x * blockDim.x + threadIdx.x;
  if (e < E) atomicAdd(&deg[dst[e]], 1.0f);
}

__global__ void deg_to_dinv(float* __restrict__ deg, int n) {
  int i = blockIdx.x * blockDim.x + threadIdx.x;
  if (i < n) {
    float d = deg[i];
    deg[i] = d > 0.0f ? rsqrtf(d) : 0.0f;
  }
}

// ------------------------ embedding gather -> bf16 --------------------------
// Hbf[i, c] = bf16(emb[x[i, c/200], c%200]) for c < 4000, else 0  (ld = 6016)
__global__ void embed_gather(const int* __restrict__ x, const float* __restrict__ emb,
                             __bf16* __restrict__ H) {
  long long t = (long long)blockIdx.x * blockDim.x + threadIdx.x;
  const long long total = (long long)NN * H1P;
  if (t >= total) return;
  int i = (int)(t / H1P);
  int c = (int)(t % H1P);
  float v = 0.0f;
  if (c < K1) {
    int tok = x[i * TT + c / DD];
    v = emb[(long long)tok * DD + (c % DD)];
  }
  H[t] = f2bf(v);
}

// -------------------- weight transpose + convert -> bf16 --------------------
// WT[n, k] = bf16(W[k, n]) with zero pad for n >= Ncols or k >= K.  (ld = Kp)
__global__ void wt_convert(const float* __restrict__ W, int K, int Ncols,
                           __bf16* __restrict__ WT, int Kp, int Np) {
  long long t = (long long)blockIdx.x * blockDim.x + threadIdx.x;
  const long long total = (long long)Np * Kp;
  if (t >= total) return;
  int n = (int)(t / Kp);
  int k = (int)(t % Kp);
  float v = (n < Ncols && k < K) ? W[(long long)k * Ncols + n] : 0.0f;
  WT[t] = f2bf(v);
}

// ------------------------------ WMMA GEMM -----------------------------------
// C[M, Np] = A[M, Kp] * BT[Np, Kp]^T ; bf16 in, f32 out. One wave per 64x64
// tile (4x4 sub-tiles of 16x16), K in steps of 32.
// A fragment (16x32, MxK): lane<16 -> K pairs {0..7,16..23}; lane>=16 -> {8..15,24..31}
// B fragment (32x16, KxN): lane<16 -> K {0..7,16..23} of column lane; lane>=16 -> {8..15,24..31}
__global__ __launch_bounds__(32)
void wmma_gemm(const __bf16* __restrict__ A, int lda,
               const __bf16* __restrict__ BT, int ldk,
               float* __restrict__ C, int ldc, int Kp) {
  const int lane = threadIdx.x & 31;
  const int half = (lane >= 16) ? 1 : 0;
  const int l15  = lane & 15;
  const int n0 = blockIdx.x * 64;
  const int m0 = blockIdx.y * 64;

  v8f acc[4][4] = {};

  const __bf16* arow[4];
  const __bf16* brow[4];
#pragma unroll
  for (int mt = 0; mt < 4; ++mt)
    arow[mt] = A + (long long)(m0 + mt * 16 + l15) * lda;
#pragma unroll
  for (int nt = 0; nt < 4; ++nt)
    brow[nt] = BT + (long long)(n0 + nt * 16 + l15) * ldk;

  const int akb1 = half ? 8  : 0;    // A: first 4 VGPR pairs
  const int akb2 = half ? 24 : 16;   // A: second 4 VGPR pairs
  const int bkb  = half ? 8 : 0;     // B: same K pairs as A

  for (int k = 0; k < Kp; k += 32) {
    v16bf a[4], b[4];
#pragma unroll
    for (int mt = 0; mt < 4; ++mt) {
      union { uint4 u[2]; v16bf v; } fa;
      fa.u[0] = *(const uint4*)(arow[mt] + k + akb1);
      fa.u[1] = *(const uint4*)(arow[mt] + k + akb2);
      a[mt] = fa.v;
    }
#pragma unroll
    for (int nt = 0; nt < 4; ++nt) {
      union { uint4 u[2]; v16bf v; } fb;
      fb.u[0] = *(const uint4*)(brow[nt] + k + bkb);
      fb.u[1] = *(const uint4*)(brow[nt] + k + bkb + 16);
      b[nt] = fb.v;
    }
#pragma unroll
    for (int mt = 0; mt < 4; ++mt)
#pragma unroll
      for (int nt = 0; nt < 4; ++nt)
        acc[mt][nt] = __builtin_amdgcn_wmma_f32_16x16x32_bf16(
            false, a[mt], false, b[nt], (short)0, acc[mt][nt], false, false);
  }

  // C/D layout: VGPR r -> (M = r + 8*half, N = l15) within each 16x16 tile
#pragma unroll
  for (int mt = 0; mt < 4; ++mt) {
    int mb = m0 + mt * 16 + half * 8;
#pragma unroll
    for (int nt = 0; nt < 4; ++nt) {
      int n = n0 + nt * 16 + l15;
#pragma unroll
      for (int r = 0; r < 8; ++r)
        C[(long long)(mb + r) * ldc + n] = acc[mt][nt][r];
    }
  }
}

// --------------------------- edge aggregation -------------------------------
// Z[dst, :] += dinv[src]*dinv[dst] * Y[src, :]  over E edges + N self-loops.
// One wave per edge, lanes stride the 6000 columns.
__global__ void aggregate(const int* __restrict__ ei, const float* __restrict__ dinv,
                          const float* __restrict__ Y, float* __restrict__ Z) {
  int wid  = blockIdx.x * (blockDim.x >> 5) + (threadIdx.x >> 5);
  int lane = threadIdx.x & 31;
  const int total = EE + NN;
  if (wid >= total) return;
  int s, d;
  if (wid < EE) { s = ei[wid]; d = ei[EE + wid]; }
  else          { s = d = wid - EE; }
  float norm = dinv[s] * dinv[d];
  const float* ys = Y + (long long)s * H1P;
  float* zd       = Z + (long long)d * H1P;
  for (int c = lane; c < H1; c += 32)
    atomicAdd(&zd[c], norm * ys[c]);
}

// ------------------------ bias + leaky epilogue -----------------------------
// v = leaky(Z[i,c] + bias[c]); optionally write f32 back in place and/or bf16.
__global__ void bias_act(float* __restrict__ Z, int ld, const float* __restrict__ bias,
                         int M, int Ncols, __bf16* __restrict__ Hout,
                         int write_bf16, int write_f32) {
  long long t = (long long)blockIdx.x * blockDim.x + threadIdx.x;
  const long long total = (long long)M * Ncols;
  if (t >= total) return;
  int i = (int)(t / Ncols);
  int c = (int)(t % Ncols);
  long long off = (long long)i * ld + c;
  float v = leaky(Z[off] + bias[c]);
  if (write_f32)  Z[off] = v;
  if (write_bf16) Hout[off] = f2bf(v);
}

// ------------------------------ max pool ------------------------------------
static __device__ __forceinline__ void atomicMaxF(float* addr, float val) {
  int* ia = (int*)addr;
  int old = __float_as_int(*addr);
  while (__int_as_float(old) < val) {
    int assumed = old;
    old = atomicCAS(ia, assumed, __float_as_int(val));
    if (old == assumed) break;
  }
}

__global__ void pool_max(const float* __restrict__ Z, const int* __restrict__ batch,
                         float* __restrict__ G) {
  long long t = (long long)blockIdx.x * blockDim.x + threadIdx.x;
  const long long total = (long long)NN * H1;
  if (t >= total) return;
  int i = (int)(t / H1);
  int c = (int)(t % H1);
  atomicMaxF(&G[(long long)batch[i] * H1P + c], Z[(long long)i * H1P + c]);
}

__global__ void g_to_bf(const float* __restrict__ G, __bf16* __restrict__ Gbf) {
  long long t = (long long)blockIdx.x * blockDim.x + threadIdx.x;
  const long long total = (long long)BB * H1P;
  if (t >= total) return;
  int c = (int)(t % H1P);
  float v = (c < H1) ? G[t] : 0.0f;
  Gbf[t] = f2bf(v);
}

// ------------------------------ final layer ---------------------------------
// out[b,o] = leaky(sum_k L1[b,k]*Wl2[k,o] + bl2[o]);  64x4 outputs, plain f32.
__global__ void final_linear(const float* __restrict__ L1, int ld,
                             const float* __restrict__ Wl2, const float* __restrict__ bl2,
                             float* __restrict__ out) {
  int t = blockIdx.x * blockDim.x + threadIdx.x;
  if (t >= BB * 4) return;
  int b = t >> 2, o = t & 3;
  float s = bl2[o];
  const float* row = L1 + (long long)b * ld;
  for (int k = 0; k < L1N; ++k)
    s += row[k] * Wl2[k * 4 + o];
  out[t] = leaky(s);
}

// ---------------------------------------------------------------------------

static inline long long cdiv(long long a, long long b) { return (a + b - 1) / b; }

extern "C" void kernel_launch(void* const* d_in, const int* in_sizes, int n_in,
                              void* d_out, int out_size, void* d_ws, size_t ws_size,
                              hipStream_t stream) {
  (void)in_sizes; (void)n_in; (void)out_size; (void)ws_size;

  const int*   x    = (const int*)  d_in[0];
  const int*   ei   = (const int*)  d_in[1];   // [2, E] row-major
  const int*   bat  = (const int*)  d_in[2];
  const float* emb  = (const float*)d_in[3];
  const float* W1   = (const float*)d_in[4];
  const float* b1   = (const float*)d_in[5];
  const float* W2   = (const float*)d_in[6];
  const float* b2   = (const float*)d_in[7];
  const float* Wl1  = (const float*)d_in[8];
  const float* bl1  = (const float*)d_in[9];
  const float* Wl2  = (const float*)d_in[10];
  const float* bl2  = (const float*)d_in[11];
  float* out = (float*)d_out;

  // --- carve workspace (256B aligned) ---
  char* w = (char*)d_ws;
  auto carve = [&](size_t bytes) -> void* {
    void* p = (void*)w;
    w += (bytes + 255) & ~(size_t)255;
    return p;
  };
  __bf16* Hbf  = (__bf16*)carve((size_t)NN  * H1P * 2);   // activations (bf16)
  __bf16* WT   = (__bf16*)carve((size_t)H1P * H1P * 2);   // transposed weights (bf16)
  float*  Y    = (float*) carve((size_t)NN  * H1P * 4);   // GEMM output
  float*  Z    = (float*) carve((size_t)NN  * H1P * 4);   // aggregation output
  float*  dinv = (float*) carve((size_t)NN * 4);
  float*  G    = (float*) carve((size_t)BB * H1P * 4);
  __bf16* Gbf  = (__bf16*)carve((size_t)BB * H1P * 2);

  const int TPB = 256;

  // ---- degrees (self-loops = 1.0 seed) -> dinv ----
  fill_f32<<<(unsigned)cdiv(NN, TPB), TPB, 0, stream>>>(dinv, NN, 1.0f);
  deg_accum<<<(unsigned)cdiv(EE, TPB), TPB, 0, stream>>>(ei + EE, EE, dinv);
  deg_to_dinv<<<(unsigned)cdiv(NN, TPB), TPB, 0, stream>>>(dinv, NN);

  // ---- embedding gather -> Hbf [8000, 6016] (cols >= 4000 zeroed) ----
  embed_gather<<<(unsigned)cdiv((long long)NN * H1P, TPB), TPB, 0, stream>>>(x, emb, Hbf);

  // ================= layer 1: Y = H @ W1 (K=4000) =================
  wt_convert<<<(unsigned)cdiv((long long)H1P * K1, TPB), TPB, 0, stream>>>(
      W1, K1, H1, WT, K1, H1P);
  wmma_gemm<<<dim3(H1P / 64, NN / 64), 32, 0, stream>>>(Hbf, H1P, WT, K1, Y, H1P, K1);

  fill_f32<<<(unsigned)cdiv((long long)NN * H1P, TPB), TPB, 0, stream>>>(
      Z, (long long)NN * H1P, 0.0f);
  aggregate<<<(unsigned)cdiv(EE + NN, TPB / 32), TPB, 0, stream>>>(ei, dinv, Y, Z);
  bias_act<<<(unsigned)cdiv((long long)NN * H1, TPB), TPB, 0, stream>>>(
      Z, H1P, b1, NN, H1, Hbf, /*bf16*/1, /*f32*/0);

  // ================= layer 2: Y = H1 @ W2 (K=6016 padded) =================
  wt_convert<<<(unsigned)cdiv((long long)H1P * H1P, TPB), TPB, 0, stream>>>(
      W2, H1, H1, WT, H1P, H1P);
  wmma_gemm<<<dim3(H1P / 64, NN / 64), 32, 0, stream>>>(Hbf, H1P, WT, H1P, Y, H1P, H1P);

  fill_f32<<<(unsigned)cdiv((long long)NN * H1P, TPB), TPB, 0, stream>>>(
      Z, (long long)NN * H1P, 0.0f);
  aggregate<<<(unsigned)cdiv(EE + NN, TPB / 32), TPB, 0, stream>>>(ei, dinv, Y, Z);
  bias_act<<<(unsigned)cdiv((long long)NN * H1, TPB), TPB, 0, stream>>>(
      Z, H1P, b2, NN, H1, ((__bf16*)0), /*bf16*/0, /*f32*/1);

  // ================= global max pool -> G [64, 6016] =================
  fill_f32<<<(unsigned)cdiv((long long)BB * H1P, TPB), TPB, 0, stream>>>(
      G, (long long)BB * H1P, -__builtin_huge_valf());
  pool_max<<<(unsigned)cdiv((long long)NN * H1, TPB), TPB, 0, stream>>>(Z, bat, G);
  g_to_bf<<<(unsigned)cdiv((long long)BB * H1P, TPB), TPB, 0, stream>>>(G, Gbf);

  // ================= lin1: Y[64, 3008] = G @ Wl1 =================
  wt_convert<<<(unsigned)cdiv((long long)L1NP * H1P, TPB), TPB, 0, stream>>>(
      Wl1, H1, L1N, WT, H1P, L1NP);
  wmma_gemm<<<dim3(L1NP / 64, BB / 64), 32, 0, stream>>>(Gbf, H1P, WT, H1P, Y, L1NP, H1P);
  bias_act<<<(unsigned)cdiv((long long)BB * L1N, TPB), TPB, 0, stream>>>(
      Y, L1NP, bl1, BB, L1N, ((__bf16*)0), /*bf16*/0, /*f32*/1);

  // ================= lin2: out[64, 4] =================
  final_linear<<<1, 256, 0, stream>>>(Y, L1NP, Wl2, bl2, out);
}
